// Gate_40956808135215
// MI455X (gfx1250) — matrix operand order repaired and checked by reading, not compiled, not run on hardware
//
#include <hip/hip_runtime.h>

// ---------------------------------------------------------------------------
// MoE router gate for MI455X (gfx1250):
//   logits = x @ W^T (fp32 accuracy via bf16x3-split WMMA), fused softmax +
//   bias + top-8 epilogue. Each wave owns 32 tokens x 64 experts; B (weight)
//   fragments are reused across two 16-row M-tiles -> 1.0 loads per WMMA.
//   Logits never touch HBM.
// ---------------------------------------------------------------------------

typedef __attribute__((ext_vector_type(16))) __bf16 v16bf;
typedef __attribute__((ext_vector_type(8)))  float  v8f;

#define DIM       2048
#define NEXP      64
#define TOPK_N    8
#define ROWSTRIDE 65     // 64 + 1 pad -> conflict-free per-token LDS rows
#define WPB       4      // waves per block (128 threads)
#define MTILE     32     // tokens per wave

__device__ __forceinline__ v8f wmma_bf16(v16bf a, v16bf b, v8f c) {
  // (neg_a, A, neg_b, B, c_mod, C, reuse_a, reuse_b)
  return __builtin_amdgcn_wmma_f32_16x16x32_bf16(false, a, false, b,
                                                 (short)0, c, false, false);
}

// Build one 16x32 bf16 A-fragment pair (hi/lo split) from fp32 row data,
// 16 contiguous K per lane (matching B's contiguous K order).
__device__ __forceinline__ void make_a(const float* __restrict__ p,
                                       v16bf& ah, v16bf& al) {
  const float4* pa = reinterpret_cast<const float4*>(p);
  const float4* pb = reinterpret_cast<const float4*>(p + 8);
  float4 c0 = pa[0], c1 = pa[1], c2 = pb[0], c3 = pb[1];
  float fa[16] = {c0.x, c0.y, c0.z, c0.w, c1.x, c1.y, c1.z, c1.w,
                  c2.x, c2.y, c2.z, c2.w, c3.x, c3.y, c3.z, c3.w};
#pragma unroll
  for (int i = 0; i < 16; ++i) {
    __bf16 h = (__bf16)fa[i];
    ah[i] = h;
    al[i] = (__bf16)(fa[i] - (float)h);
  }
}

// Split fp32 weight into bf16 hi/lo pair (one-time, 512 KB total).
__global__ void cvt_weight_kernel(const float* __restrict__ w,
                                  __bf16* __restrict__ wh,
                                  __bf16* __restrict__ wl) {
  int i = blockIdx.x * blockDim.x + threadIdx.x;
  float f = w[i];
  __bf16 h = (__bf16)f;
  wh[i] = h;
  wl[i] = (__bf16)(f - (float)h);
}

__global__ __launch_bounds__(128) void router_kernel(
    const float*  __restrict__ x,      // [tokens, DIM] fp32
    const __bf16* __restrict__ wh,     // [NEXP, DIM] bf16 hi
    const __bf16* __restrict__ wl,     // [NEXP, DIM] bf16 lo
    const float*  __restrict__ bias,   // [NEXP]
    float*        __restrict__ wout,   // [tokens, 8] fp32
    int*          __restrict__ iout,   // [tokens, 8] int32
    int tokens) {
  __shared__ float lds[WPB][MTILE * ROWSTRIDE];  // per-wave 32x64 logits tile
  __shared__ float sbias[NEXP];

  const int tid  = threadIdx.x;
  const int wv   = tid >> 5;        // wave id in block (uniform)
  const int lane = tid & 31;
  const int l16  = lane & 15;
  const int hi   = lane >> 4;       // 0: lanes 0-15, 1: lanes 16-31

  if (tid < NEXP) sbias[tid] = bias[tid];

  const int gw = blockIdx.x * WPB + wv;   // global wave id
  const int m0 = gw * MTILE;              // first token of this wave's tile

  // A: lane<16 owns K {0..15}, lane>=16 owns
  // K {16..31}. Two M-tiles: rows m0+l16 and m0+16+l16.
  const float* xrow0 = x + (size_t)(m0 + l16) * DIM + hi * 16;
  const float* xrow1 = xrow0 + (size_t)16 * DIM;
  // B: contiguous K order, same as A. lane n: N = n&15, K offset (n>=16)*16,
  // 16 contiguous bf16 per lane per K-step.
  const __bf16* bh0 = wh + (size_t)l16 * DIM + hi * 16;
  const __bf16* bl0 = wl + (size_t)l16 * DIM + hi * 16;

  v8f acc[8] = {};  // [0..3]: tokens 0-15, [4..7]: tokens 16-31 (4 N-tiles)

  for (int k0 = 0; k0 < DIM; k0 += 32) {
    v16bf a0h, a0l, a1h, a1l;
    make_a(xrow0 + k0, a0h, a0l);
    make_a(xrow1 + k0, a1h, a1l);
#pragma unroll
    for (int j = 0; j < 4; ++j) {
      const v16bf b_hi = *reinterpret_cast<const v16bf*>(bh0 + (size_t)j * 16 * DIM + k0);
      const v16bf b_lo = *reinterpret_cast<const v16bf*>(bl0 + (size_t)j * 16 * DIM + k0);
      // 3-term split, two M-tiles sharing each B fragment
      acc[j]     = wmma_bf16(a0h, b_hi, acc[j]);
      acc[4 + j] = wmma_bf16(a1h, b_hi, acc[4 + j]);
      acc[j]     = wmma_bf16(a0l, b_hi, acc[j]);
      acc[4 + j] = wmma_bf16(a1l, b_hi, acc[4 + j]);
      acc[j]     = wmma_bf16(a0h, b_lo, acc[j]);
      acc[4 + j] = wmma_bf16(a1h, b_lo, acc[4 + j]);
    }
  }

  // ---- spill logits tiles to LDS (C layout: VGPR r, lane half -> row) ----
#pragma unroll
  for (int j = 0; j < 4; ++j) {
#pragma unroll
    for (int r = 0; r < 8; ++r) {
      int trow = hi ? (r + 8) : r;
      lds[wv][trow * ROWSTRIDE + j * 16 + l16]        = acc[j][r];
      lds[wv][(trow + 16) * ROWSTRIDE + j * 16 + l16] = acc[4 + j][r];
    }
  }
  __syncthreads();

  // ---- fused epilogue: softmax, bias, top-8 (one token per lane) ----
  {
    float* row = &lds[wv][lane * ROWSTRIDE];
    float m = -__builtin_inff();
    for (int i = 0; i < NEXP; ++i) m = fmaxf(m, row[i]);
    float s = 0.f;
    for (int i = 0; i < NEXP; ++i) {
      float e = __expf(row[i] - m);
      row[i] = e;            // overwrite logits with exp values
      s += e;
    }
    const float inv = 1.f / s;
    const int token = m0 + lane;
    for (int j = 0; j < TOPK_N; ++j) {
      float best = -__builtin_inff();
      int bi = 0;
      for (int i = 0; i < NEXP; ++i) {
        float v = row[i] * inv + sbias[i];    // biased score
        if (v > best) { best = v; bi = i; }   // strict > : ties -> lower index
      }
      wout[token * TOPK_N + j] = row[bi] * inv;  // unbiased softmax weight
      iout[token * TOPK_N + j] = bi;
      row[bi] = -__builtin_inff();               // exclude from later passes
    }
  }
}

extern "C" void kernel_launch(void* const* d_in, const int* in_sizes, int n_in,
                              void* d_out, int out_size, void* d_ws, size_t ws_size,
                              hipStream_t stream) {
  const float* x    = (const float*)d_in[0];  // [tokens, 2048]
  const float* w    = (const float*)d_in[1];  // [64, 2048]
  const float* bias = (const float*)d_in[2];  // [64]
  const int tokens  = in_sizes[0] / DIM;      // 16384

  // workspace: bf16 hi/lo weight copies (2 * 64*2048*2 B = 512 KB)
  __bf16* wh = (__bf16*)d_ws;
  __bf16* wl = wh + (size_t)NEXP * DIM;
  cvt_weight_kernel<<<(NEXP * DIM) / 256, 256, 0, stream>>>(w, wh, wl);

  float* wout = (float*)d_out;
  int*   iout = (int*)(wout + (size_t)tokens * TOPK_N);

  // 32 tokens per wave, 4 waves per block -> tokens/128 blocks
  router_kernel<<<tokens / (WPB * MTILE), WPB * 32, 0, stream>>>(
      x, wh, wl, bias, wout, iout, tokens);
}
